// HelmBlock_31009663877808
// MI455X (gfx1250) — matrix operand order repaired and checked by reading, not compiled, not run on hardware
//
#include <hip/hip_runtime.h>
#include <hip/hip_bf16.h>
#include <math.h>

typedef __attribute__((ext_vector_type(16))) _Float16 v16h;
typedef __attribute__((ext_vector_type(8)))  _Float16 v8h;
typedef __attribute__((ext_vector_type(8)))  float    v8f;

union V16 { v16h v; v8h h[2]; };

static __device__ __forceinline__ v8f wmma_f32_f16(v16h a, v16h b, v8f c) {
  return __builtin_amdgcn_wmma_f32_16x16x32_f16(false, a, false, b, (short)0, c, false, false);
}
// contiguous 32B per-lane chunk -> two b128 loads
static __device__ __forceinline__ v16h ld_pk(const _Float16* p) {
  V16 u;
  u.h[0] = *(const v8h*)(p);
  u.h[1] = *(const v8h*)(p + 8);
  return u.v;
}

#define HS 256
#define HE 64
#define CENC 64
#define KENC 128
#define NCORR 162
#define NCP 176      // NCORR padded to mult of 16
#define EG 16
#define HT 252
#define NBG 16

// d_out regions (float offsets), return order: x, helm, vel, vel_phi, vel_vort
#define OFF_X    ((size_t)0)
#define OFF_HELM ((size_t)2*128*252*252)
#define OFF_VEL  (OFF_HELM + (size_t)16*2*256*256)
#define OFF_VPHI (OFF_VEL  + (size_t)16*2*252*252)
#define OFF_VVOR (OFF_VPHI + (size_t)16*2*252*252)

// ---------------- weight pre-pack into WMMA B-operand order ----------------
// out[i], i = ((slot*32)+lane)*16 + e ; k = slot_k*32 + 8*hi + (e&7) + 16*(e>>3) ; oc/n = nloc (+16*nt)
__global__ void k_pack(const float* __restrict__ w, _Float16* __restrict__ out,
                       int mode, int total)
{
  int i = blockIdx.x * blockDim.x + threadIdx.x;
  if (i >= total) return;
  int e = i & 15;
  int lane = (i >> 4) & 31;
  int slot = i >> 9;
  int hi = lane >> 4, nloc = lane & 15;
  float val = 0.f;
  if (mode == 0) {               // encoder: slot = nt*4 + ks, K = ic*16+ky*4+kx
    int nt = slot >> 2, ks = slot & 3;
    int k = ks * 32 + 8 * hi + (e & 7) + 16 * (e >> 3);
    val = w[(nt * 16 + nloc) * KENC + k];
  } else if (mode == 1) {        // conv1: slot = ks (0..50, slot 50 = zero pad), k = tap*176 + icp
    int k = slot * 32 + 8 * hi + (e & 7) + 16 * (e >> 3);
    int tap = k / NCP, icp = k - tap * NCP;
    if (tap < 9 && icp < NCORR) val = w[nloc * 1458 + icp * 9 + tap];
  } else if (mode == 2) {        // conv2: slot = ks (0..4), k = tap*16 + ic
    int k = slot * 32 + 8 * hi + (e & 7) + 16 * (e >> 3);
    if (k < 144) val = w[nloc * 144 + (k & 15) * 9 + (k >> 4)];
  } else {                       // mlp: slot = nt*4 + ks, w is [k][n]
    int nt = slot >> 2, ks = slot & 3;
    int k = ks * 32 + 8 * hi + (e & 7) + 16 * (e >> 3);
    val = w[k * 128 + (nt * 16 + nloc)];
  }
  out[i] = (_Float16)val;
}

// ---------------- encoder: patchify conv as WMMA GEMM, A and B in registers ----------------
// out layout: (bg, y, x, c=64) f32
__global__ void __launch_bounds__(128)
k_encode(const float* __restrict__ in, const float* __restrict__ map,
         const _Float16* __restrict__ wpk, const float* __restrict__ bias,
         float* __restrict__ out)
{
  const int tid = threadIdx.x, lane = tid & 31, wave = tid >> 5;
  const int y = blockIdx.x, bg = blockIdx.y;
  const int b = bg >> 3, g = bg & 7;
  const int x0 = wave * 16;
  const int m = lane & 15, hi = lane >> 4, nloc = lane & 15;
  // Build A (16x32 tile rows) directly in registers: k-groups map to 2 contiguous float4 rows
  v16h A[4];
#pragma unroll
  for (int ks = 0; ks < 4; ++ks) {
    V16 u;
#pragma unroll
    for (int grp = 0; grp < 2; ++grp) {
      int s = ks * 32 + 8 * hi + 16 * grp;   // 8-aligned group start; same ic for 8 elems
      int ic = s >> 4;
      int ky = (s >> 2) & 3;
      int Y0 = y * 4 + ky;
      int X = (x0 + m) * 4;
      const float4 a0 = *(const float4*)&in[(((size_t)b * 64 + (g * 8 + ic)) * HS + Y0) * HS + X];
      const float4 a1 = *(const float4*)&in[(((size_t)b * 64 + (g * 8 + ic)) * HS + Y0 + 1) * HS + X];
      const float4 m0 = *(const float4*)&map[((size_t)b * HS + Y0) * HS + X];
      const float4 m1 = *(const float4*)&map[((size_t)b * HS + Y0 + 1) * HS + X];
      v8h hh;
      hh[0] = (_Float16)(a0.x * m0.x); hh[1] = (_Float16)(a0.y * m0.y);
      hh[2] = (_Float16)(a0.z * m0.z); hh[3] = (_Float16)(a0.w * m0.w);
      hh[4] = (_Float16)(a1.x * m1.x); hh[5] = (_Float16)(a1.y * m1.y);
      hh[6] = (_Float16)(a1.z * m1.z); hh[7] = (_Float16)(a1.w * m1.w);
      u.h[grp] = hh;
    }
    A[ks] = u.v;
  }
  // Preload all 16 B tiles into distinct registers, then pure WMMA work
  v16h Bv[16];
#pragma unroll
  for (int nt = 0; nt < 4; ++nt)
#pragma unroll
    for (int ks = 0; ks < 4; ++ks)
      Bv[nt * 4 + ks] = ld_pk(&wpk[((size_t)(nt * 4 + ks) * 32 + lane) * 16]);
  v8f acc[4];
  { v8f z = {}; for (int nt = 0; nt < 4; ++nt) acc[nt] = z; }
#pragma unroll
  for (int ks = 0; ks < 4; ++ks)
#pragma unroll
    for (int nt = 0; nt < 4; ++nt)
      acc[nt] = wmma_f32_f16(A[ks], Bv[nt * 4 + ks], acc[nt]);
#pragma unroll
  for (int nt = 0; nt < 4; ++nt)
#pragma unroll
    for (int r = 0; r < 8; ++r) {
      int oc = nt * 16 + nloc;
      int mm = r + 8 * hi;
      out[(((size_t)bg * HE + y) * HE + (x0 + mm)) * CENC + oc] = acc[nt][r] + bias[oc];
    }
}

// ---------------- correlation ----------------
__global__ void k_correlation(const float* __restrict__ fpm, const float* __restrict__ fpb,
                              const float* __restrict__ fnm, _Float16* __restrict__ corr)
{
  size_t idx = (size_t)blockIdx.x * blockDim.x + threadIdx.x;
  const size_t total = (size_t)NBG * HE * HE * NCORR;
  if (idx >= total) return;
  int c = (int)(idx % NCORR);
  size_t t = idx / NCORR;
  int x = (int)(t % HE); t /= HE;
  int y = (int)(t % HE); int bg = (int)(t / HE);
  const float* f1 = (c < 81) ? fpm : fpb;
  int d = (c < 81) ? c : c - 81;
  int dy = d / 9 - 4, dx = d % 9 - 4;
  int ys = y + dy, xs = x + dx;
  float s = 0.f;
  if (ys >= 0 && ys < HE && xs >= 0 && xs < HE) {
    const float* a  = f1  + (((size_t)bg * HE + y ) * HE + x ) * CENC;
    const float* bb = fnm + (((size_t)bg * HE + ys) * HE + xs) * CENC;
    for (int ch = 0; ch < CENC; ++ch) s += a[ch] * bb[ch];
    s *= (1.f / 64.f);
  }
  corr[idx] = (_Float16)s;
}

// ---------------- conv1: up2x(corr) -> 3x3 conv 162->16, bn_relu ----------------
// 2 waves (32 px) per block; K tap-major padded to 1600 (50 ksteps);
// B double-buffered in registers, streamed packed from L2
__global__ void __launch_bounds__(64)
k_conv1(const _Float16* __restrict__ corr, const _Float16* __restrict__ wpk,
        const float* __restrict__ gam, const float* __restrict__ bet,
        _Float16* __restrict__ h1)
{
  __shared__ _Float16 sIn[3 * 34 * NCP];   // 35904 B, channel-padded to 176
  const int tid = threadIdx.x, lane = tid & 31, wave = tid >> 5;
  const int px0 = blockIdx.x * 32;
  const int uy = blockIdx.y;
  const int bg = blockIdx.z;
  for (int i = tid; i < 3 * 34 * NCP; i += 64) {
    int ch = i % NCP;
    int t = i / NCP;
    int cc = t % 34, r = t / 34;
    int uyy = uy - 1 + r, uxx = px0 - 1 + cc;
    float v = 0.f;
    if (ch < NCORR && uyy >= 0 && uyy < 128 && uxx >= 0 && uxx < 128) {
      float fy = uyy * (63.f / 127.f), fx = uxx * (63.f / 127.f);
      int y0 = (int)fy, x0 = (int)fx;
      int y1 = min(y0 + 1, 63), x1 = min(x0 + 1, 63);
      float wy = fy - y0, wx = fx - x0;
      const _Float16* base = corr + ((size_t)bg * HE * HE) * NCORR;
      float c00 = (float)base[((y0 * HE) + x0) * NCORR + ch];
      float c01 = (float)base[((y0 * HE) + x1) * NCORR + ch];
      float c10 = (float)base[((y1 * HE) + x0) * NCORR + ch];
      float c11 = (float)base[((y1 * HE) + x1) * NCORR + ch];
      v = (c00 * (1 - wx) + c01 * wx) * (1 - wy) + (c10 * (1 - wx) + c11 * wx) * wy;
    }
    sIn[i] = (_Float16)v;
  }
  __syncthreads();
  v8f acc = {};
  const int m = lane & 15, hi = lane >> 4, nloc = lane & 15;
  const v8h zero8 = {};
  v16h bcur = ld_pk(&wpk[(size_t)lane * 16]);      // ks = 0
  for (int ks = 0; ks < 50; ++ks) {
    v16h bnxt = ld_pk(&wpk[((size_t)(ks + 1) * 32 + lane) * 16]);   // slot 50 = zeros
    int kb = ks * 32;
    V16 ua;
#pragma unroll
    for (int grp = 0; grp < 2; ++grp) {
      int s = kb + 8 * hi + 16 * grp;        // 8-aligned; tap & icp constant in group
      v8h hh = zero8;
      if (s < 1584) {
        int tap = s / NCP, icp = s - tap * NCP;
        int r = tap / 3, c = tap - r * 3;
        hh = *(const v8h*)&sIn[((r * 34 + wave * 16 + m + c) * NCP) + icp];
      }
      ua.h[grp] = hh;
    }
    acc = wmma_f32_f16(ua.v, bcur, acc);
    bcur = bnxt;
  }
  const float rs = rsqrtf(1.00001f);
  float sc = gam[nloc] * rs;
  float bo = bet[nloc];
#pragma unroll
  for (int r = 0; r < 8; ++r) {
    int mm = r + 8 * hi;
    float v = fmaxf(acc[r] * sc + bo, 0.f);
    h1[(((size_t)bg * 128 + uy) * 128 + (px0 + wave * 16 + mm)) * EG + nloc] = (_Float16)v;
  }
}

// ---------------- conv2: up2x(h1) -> 3x3 conv 16->16, bn_relu, fused 1x1 -> helm ----------------
__global__ void __launch_bounds__(64)
k_conv2(const _Float16* __restrict__ h1, const _Float16* __restrict__ wpk,
        const float* __restrict__ gam, const float* __restrict__ bet,
        const float* __restrict__ ow, const float* __restrict__ ob,
        const float* __restrict__ wsc, float extra, int ch,
        float* __restrict__ helm)
{
  __shared__ _Float16 sIn[3 * 34 * EG];
  __shared__ float sD[2][16][16];
  const int tid = threadIdx.x, lane = tid & 31, wave = tid >> 5;
  const int px0b = blockIdx.x * 32;
  const int uy = blockIdx.y; const int bg = blockIdx.z;
  for (int i = tid; i < 3 * 34 * EG; i += 64) {
    int c16 = i & 15;
    int t = i >> 4;
    int cc = t % 34, r = t / 34;
    int uyy = uy - 1 + r, uxx = px0b - 1 + cc;
    float v = 0.f;
    if (uyy >= 0 && uyy < 256 && uxx >= 0 && uxx < 256) {
      float fy = uyy * (127.f / 255.f), fx = uxx * (127.f / 255.f);
      int y0 = (int)fy, x0 = (int)fx; int y1 = min(y0 + 1, 127), x1 = min(x0 + 1, 127);
      float wy = fy - y0, wx = fx - x0;
      const _Float16* base = h1 + ((size_t)bg * 128 * 128) * EG;
      float c00 = (float)base[((y0 * 128) + x0) * EG + c16];
      float c01 = (float)base[((y0 * 128) + x1) * EG + c16];
      float c10 = (float)base[((y1 * 128) + x0) * EG + c16];
      float c11 = (float)base[((y1 * 128) + x1) * EG + c16];
      v = (c00 * (1 - wx) + c01 * wx) * (1 - wy) + (c10 * (1 - wx) + c11 * wx) * wy;
    }
    sIn[i] = (_Float16)v;
  }
  __syncthreads();
  v8f acc = {};
  const int m = lane & 15, hi = lane >> 4, nloc = lane & 15;
  const v8h zero8 = {};
  v16h Bv[5];
#pragma unroll
  for (int ks = 0; ks < 5; ++ks) Bv[ks] = ld_pk(&wpk[((size_t)ks * 32 + lane) * 16]);
#pragma unroll
  for (int ks = 0; ks < 5; ++ks) {
    int kb = ks * 32;
    V16 ua;
#pragma unroll
    for (int grp = 0; grp < 2; ++grp) {
      int s = kb + 8 * hi + 16 * grp;        // k = tap*16 + ic
      v8h hh = zero8;
      if (s < 144) {
        int tap = s >> 4, ic0 = s & 15;
        int r = tap / 3, c = tap - r * 3;
        hh = *(const v8h*)&sIn[((r * 34 + wave * 16 + m + c) * EG) + ic0];
      }
      ua.h[grp] = hh;
    }
    acc = wmma_f32_f16(ua.v, Bv[ks], acc);
  }
#pragma unroll
  for (int r = 0; r < 8; ++r) sD[wave][r + 8 * hi][nloc] = acc[r];
  __syncthreads();
  if (lane < 16) {
    const float rs = rsqrtf(1.00001f);
    float sum = 0.f;
    for (int oc = 0; oc < 16; ++oc) {
      float v = fmaxf(sD[wave][lane][oc] * (gam[oc] * rs) + bet[oc], 0.f);
      sum += v * ow[oc];
    }
    float res = (sum + ob[0]) * (wsc[0] * extra);
    int ux = px0b + wave * 16 + lane;
    helm[(((size_t)bg * 2 + ch) * 256 + uy) * 256 + ux] = res;
  }
}

// ---------------- velocities from helm ----------------
__global__ void k_velocity(float* __restrict__ dout)
{
  size_t idx = (size_t)blockIdx.x * blockDim.x + threadIdx.x;
  const size_t total = (size_t)NBG * HT * HT;
  if (idx >= total) return;
  int x = (int)(idx % HT); size_t t = idx / HT;
  int y = (int)(t % HT); int bg = (int)(t / HT);
  const float* H = dout + OFF_HELM;
#define PHI_AT(yy, xx) H[(((size_t)bg * 2 + 0) * 256 + (yy)) * 256 + (xx)]
  float up = 0.5f * (PHI_AT(y + 2, x + 3) - PHI_AT(y + 2, x + 1));
  float vp = 0.5f * (PHI_AT(y + 3, x + 2) - PHI_AT(y + 1, x + 2));
  const float diffk[5] = {0.f, -0.5f, 0.f, 0.5f, 0.f};
  const float smok[5]  = {0.f, 0.25f, 0.5f, 0.25f, 0.f};
  auto WV = [&](int i, int j) -> float {
    if (i < 0 || i > 254 || j < 0 || j > 254) return 0.f;
    return H[(((size_t)bg * 2 + 1) * 256 + i) * 256 + j];
  };
  auto C0 = [&](int R, int C) -> float {
    float s = 0.f;
    for (int a2 = 1; a2 <= 3; ++a2)
      for (int b2 = 1; b2 <= 3; ++b2)
        s += diffk[a2] * smok[b2] * WV(R - a2, C - b2);
    return s;
  };
  auto C1 = [&](int R, int C) -> float {
    float s = 0.f;
    for (int a2 = 1; a2 <= 3; ++a2)
      for (int b2 = 1; b2 <= 3; ++b2)
        s += -smok[a2] * diffk[b2] * WV(R - a2, C - b2);
    return s;
  };
  float uv = 0.5f * (C0(y + 3, x + 2) + C0(y + 3, x + 3));
  float vv = 0.5f * (C1(y + 2, x + 3) + C1(y + 3, x + 3));
  size_t o0 = (((size_t)bg * 2 + 0) * HT + y) * HT + x;
  size_t o1 = (((size_t)bg * 2 + 1) * HT + y) * HT + x;
  dout[OFF_VPHI + o0] = up; dout[OFF_VPHI + o1] = vp;
  dout[OFF_VVOR + o0] = uv; dout[OFF_VVOR + o1] = vv;
  dout[OFF_VEL  + o0] = up + uv; dout[OFF_VEL + o1] = vp + vv;
}

// ---------------- backward bilinear warp (BFECC passes) ----------------
__global__ void k_warp(const float* __restrict__ src, const float* __restrict__ vel,
                       float velsgn, const float* __restrict__ mixsrc,
                       float mixa, float mixb, float* __restrict__ out)
{
  int idx = blockIdx.x * blockDim.x + threadIdx.x;
  if (idx >= HT * HT) return;
  int bg = blockIdx.y;
  int x = idx % HT, y = idx / HT;
  float vx = vel[(((size_t)bg * 2 + 0) * HT + y) * HT + x] * velsgn;
  float vy = vel[(((size_t)bg * 2 + 1) * HT + y) * HT + x] * velsgn;
  float px = fminf(fmaxf((float)x - vx, 0.f), 251.f);
  float py = fminf(fmaxf((float)y - vy, 0.f), 251.f);
  int x0 = (int)floorf(px), y0 = (int)floorf(py);
  int x1 = min(x0 + 1, 251), y1 = min(y0 + 1, 251);
  float wx = px - x0, wy = py - y0;
  for (int c = 0; c < 16; ++c) {
    const float* F = src + (((size_t)bg * 16 + c) * HT) * HT;
    float v00 = F[y0 * HT + x0], v01 = F[y0 * HT + x1];
    float v10 = F[y1 * HT + x0], v11 = F[y1 * HT + x1];
    float s = (v00 * (1 - wx) + v01 * wx) * (1 - wy) + (v10 * (1 - wx) + v11 * wx) * wy;
    size_t o = (((size_t)bg * 16 + c) * HT + y) * HT + x;
    float base = mixsrc ? mixsrc[o] : 0.f;
    out[o] = mixa * base + mixb * s;
  }
}

// ---------------- per-pixel LayerNorm + 2x 128x128 WMMA MLP, in-place on pred ----------------
// activations stored in LDS already in packed WMMA-A operand order
__global__ void __launch_bounds__(128)
k_mlp(float* __restrict__ xio,
      const float* __restrict__ lng, const float* __restrict__ lnb,
      const _Float16* __restrict__ w1pk, const float* __restrict__ b1,
      const _Float16* __restrict__ w2pk, const float* __restrict__ b2)
{
  __shared__ _Float16 sZ[4][4][32][16];   // [wave][ks][lane][e], 16 KB
  const int tid = threadIdx.x, lane = tid & 31, wave = tid >> 5;
  const int x0 = blockIdx.x * 64, y = blockIdx.y, b = blockIdx.z;
  const int p = lane & 15, hs = lane >> 4, nloc = lane & 15, hi = hs;
  const int xg = x0 + wave * 16 + p;
  const bool valid = (xg < HT);
  float vals[64];
  float s = 0.f, sq = 0.f;
#pragma unroll
  for (int j = 0; j < 64; ++j) {
    float v = 0.f;
    if (valid) v = xio[(((size_t)b * 128 + hs * 64 + j) * HT + y) * HT + xg];
    vals[j] = v; s += v; sq += v * v;
  }
  float mu = (s + __shfl_xor(s, 16, 32)) * (1.f / 128.f);
  float va = (sq + __shfl_xor(sq, 16, 32)) * (1.f / 128.f) - mu * mu;
  float rstd = rsqrtf(va + 1e-5f);
#pragma unroll
  for (int j = 0; j < 64; ++j) {
    int c = hs * 64 + j;
    float zn = (vals[j] - mu) * rstd * lng[c] + lnb[c];
    int rem = c & 31;
    int h2 = (rem >> 3) & 1;
    int e = (rem & 7) + ((rem >= 16) ? 8 : 0);
    sZ[wave][c >> 5][(h2 << 4) | p][e] = valid ? (_Float16)zn : (_Float16)0.f;
  }
  __syncthreads();
  v8f acc[8];
  { v8f z = {}; for (int nt = 0; nt < 8; ++nt) acc[nt] = z; }
#pragma unroll
  for (int ks = 0; ks < 4; ++ks) {
    v16h a = ld_pk(&sZ[wave][ks][lane][0]);
    v16h Bv[8];
#pragma unroll
    for (int nt = 0; nt < 8; ++nt)
      Bv[nt] = ld_pk(&w1pk[((size_t)(nt * 4 + ks) * 32 + lane) * 16]);
#pragma unroll
    for (int nt = 0; nt < 8; ++nt)
      acc[nt] = wmma_f32_f16(a, Bv[nt], acc[nt]);
  }
  __syncthreads();
#pragma unroll
  for (int nt = 0; nt < 8; ++nt)
#pragma unroll
    for (int r = 0; r < 8; ++r) {
      int n = nt * 16 + nloc, mm = r + 8 * hi;
      float v = acc[nt][r] + b1[n];
      float g = 0.5f * v * (1.f + erff(v * 0.70710678118f));
      int rem = n & 31;
      int h2 = (rem >> 3) & 1;
      int e = (rem & 7) + ((rem >= 16) ? 8 : 0);
      sZ[wave][n >> 5][(h2 << 4) | mm][e] = (_Float16)g;
    }
  __syncthreads();
  { v8f z = {}; for (int nt = 0; nt < 8; ++nt) acc[nt] = z; }
#pragma unroll
  for (int ks = 0; ks < 4; ++ks) {
    v16h a = ld_pk(&sZ[wave][ks][lane][0]);
    v16h Bv[8];
#pragma unroll
    for (int nt = 0; nt < 8; ++nt)
      Bv[nt] = ld_pk(&w2pk[((size_t)(nt * 4 + ks) * 32 + lane) * 16]);
#pragma unroll
    for (int nt = 0; nt < 8; ++nt)
      acc[nt] = wmma_f32_f16(a, Bv[nt], acc[nt]);
  }
#pragma unroll
  for (int nt = 0; nt < 8; ++nt)
#pragma unroll
    for (int r = 0; r < 8; ++r) {
      int n = nt * 16 + nloc, mm = r + 8 * hi;
      int xx = x0 + wave * 16 + mm;
      if (xx < HT) {
        size_t o = (((size_t)b * 128 + n) * HT + y) * HT + xx;
        xio[o] = xio[o] + acc[nt][r] + b2[n];
      }
    }
}

extern "C" void kernel_launch(void* const* d_in, const int* in_sizes, int n_in,
                              void* d_out, int out_size, void* d_ws, size_t ws_size,
                              hipStream_t stream) {
  (void)in_sizes; (void)n_in; (void)out_size; (void)ws_size;
  const float* prev     = (const float*)d_in[0];
  const float* nxt      = (const float*)d_in[1];
  const float* texture  = (const float*)d_in[2];
  const float* mask     = (const float*)d_in[3];
  const float* boundary = (const float*)d_in[4];
  const float* enc_w    = (const float*)d_in[5];
  const float* enc_b    = (const float*)d_in[6];
  const float* phi_w1   = (const float*)d_in[7];
  const float* phi_g1   = (const float*)d_in[8];
  const float* phi_b1   = (const float*)d_in[9];
  const float* phi_w2   = (const float*)d_in[10];
  const float* phi_g2   = (const float*)d_in[11];
  const float* phi_b2   = (const float*)d_in[12];
  const float* phi_ow   = (const float*)d_in[13];
  const float* phi_ob   = (const float*)d_in[14];
  const float* vort_w1  = (const float*)d_in[15];
  const float* vort_g1  = (const float*)d_in[16];
  const float* vort_b1  = (const float*)d_in[17];
  const float* vort_w2  = (const float*)d_in[18];
  const float* vort_g2  = (const float*)d_in[19];
  const float* vort_b2  = (const float*)d_in[20];
  const float* vort_ow  = (const float*)d_in[21];
  const float* vort_ob  = (const float*)d_in[22];
  const float* phi_wt   = (const float*)d_in[23];
  const float* vort_wt  = (const float*)d_in[24];
  const float* ln_g     = (const float*)d_in[25];
  const float* ln_b     = (const float*)d_in[26];
  const float* ff_w1    = (const float*)d_in[27];
  const float* ff_b1    = (const float*)d_in[28];
  const float* ff_w2    = (const float*)d_in[29];
  const float* ff_b2    = (const float*)d_in[30];

  float* out = (float*)d_out;

  // workspace layout
  float* fpm = (float*)d_ws;                      // 16*64*64*64 f32
  float* fpb = fpm + 4194304;
  float* fnm = fpb + 4194304;
  _Float16* corr = (_Float16*)(fnm + 4194304);    // 16*64*64*162 f16
  _Float16* h1p  = corr + 10616832;               // 16*128*128*16 f16
  _Float16* h1v  = h1p + 4194304;
  float* f1buf = (float*)(h1v + 4194304);         // 16*16*252*252 f32
  float* gbuf  = f1buf + 16257024;
  _Float16* encpk = (_Float16*)(gbuf + 16257024); // 8192
  _Float16* c1pkp = encpk + 8192;                 // 51*32*16 = 26112 (last slot zeros)
  _Float16* c1pkv = c1pkp + 26112;
  _Float16* c2pkp = c1pkv + 26112;                // 5*32*16 = 2560
  _Float16* c2pkv = c2pkp + 2560;
  _Float16* w1pk  = c2pkv + 2560;                 // 32*32*16 = 16384
  _Float16* w2pk  = w1pk + 16384;

  // 0) pre-pack all weight matrices into WMMA B-operand order (L2-resident)
  hipLaunchKernelGGL(k_pack, dim3(32),  dim3(256), 0, stream, enc_w,   encpk, 0, 8192);
  hipLaunchKernelGGL(k_pack, dim3(102), dim3(256), 0, stream, phi_w1,  c1pkp, 1, 26112);
  hipLaunchKernelGGL(k_pack, dim3(102), dim3(256), 0, stream, vort_w1, c1pkv, 1, 26112);
  hipLaunchKernelGGL(k_pack, dim3(10),  dim3(256), 0, stream, phi_w2,  c2pkp, 2, 2560);
  hipLaunchKernelGGL(k_pack, dim3(10),  dim3(256), 0, stream, vort_w2, c2pkv, 2, 2560);
  hipLaunchKernelGGL(k_pack, dim3(64),  dim3(256), 0, stream, ff_w1,   w1pk,  3, 16384);
  hipLaunchKernelGGL(k_pack, dim3(64),  dim3(256), 0, stream, ff_w2,   w2pk,  3, 16384);

  // 1) encoder (three masked tensors)
  hipLaunchKernelGGL(k_encode, dim3(64, 16), dim3(128), 0, stream, prev, mask,     encpk, enc_b, fpm);
  hipLaunchKernelGGL(k_encode, dim3(64, 16), dim3(128), 0, stream, prev, boundary, encpk, enc_b, fpb);
  hipLaunchKernelGGL(k_encode, dim3(64, 16), dim3(128), 0, stream, nxt,  mask,     encpk, enc_b, fnm);

  // 2) correlation
  {
    size_t total = (size_t)NBG * HE * HE * NCORR;
    hipLaunchKernelGGL(k_correlation, dim3((unsigned)((total + 255) / 256)), dim3(256), 0, stream,
                       fpm, fpb, fnm, corr);
  }

  // 3) double_conv_up stage 1 (both branches)
  hipLaunchKernelGGL(k_conv1, dim3(4, 128, 16), dim3(64), 0, stream, corr, c1pkp, phi_g1,  phi_b1,  h1p);
  hipLaunchKernelGGL(k_conv1, dim3(4, 128, 16), dim3(64), 0, stream, corr, c1pkv, vort_g1, vort_b1, h1v);

  // 4) stage 2 + fused 1x1 output conv -> helm
  hipLaunchKernelGGL(k_conv2, dim3(8, 256, 16), dim3(64), 0, stream,
                     h1p, c2pkp, phi_g2, phi_b2, phi_ow, phi_ob, phi_wt, 1.0f, 0, out + OFF_HELM);
  hipLaunchKernelGGL(k_conv2, dim3(8, 256, 16), dim3(64), 0, stream,
                     h1v, c2pkv, vort_g2, vort_b2, vort_ow, vort_ob, vort_wt, 256.0f, 1, out + OFF_HELM);

  // 5) velocities
  {
    size_t total = (size_t)NBG * HT * HT;
    hipLaunchKernelGGL(k_velocity, dim3((unsigned)((total + 255) / 256)), dim3(256), 0, stream, out);
  }

  // 6) BFECC warp
  const float* vel = out + OFF_VEL;
  unsigned wb = (HT * HT + 255) / 256;
  hipLaunchKernelGGL(k_warp, dim3(wb, 16), dim3(256), 0, stream,
                     texture, vel,  1.0f, (const float*)nullptr, 0.f,  1.0f, f1buf);
  hipLaunchKernelGGL(k_warp, dim3(wb, 16), dim3(256), 0, stream,
                     f1buf,   vel, -1.0f, texture,              1.5f, -0.5f, gbuf);
  hipLaunchKernelGGL(k_warp, dim3(wb, 16), dim3(256), 0, stream,
                     gbuf,    vel,  1.0f, (const float*)nullptr, 0.f,  1.0f, out + OFF_X);

  // 7) LayerNorm + MLP, in place on pred -> x
  hipLaunchKernelGGL(k_mlp, dim3(4, 252, 2), dim3(128), 0, stream,
                     out + OFF_X, ln_g, ln_b, w1pk, ff_b1, w2pk, ff_b2);
}
